// KGLayer_16630113370617
// MI455X (gfx1250) — compile-verified, matrix operand order checked by reading, not run on hardware
//
#include <hip/hip_runtime.h>
#include <hip/hip_bf16.h>
#include <math.h>

// ---------------- problem constants ----------------
#define N_ENT   100000
#define N_REL   200
#define IN_DIM  128
#define OUT_DIM 200
#define NEDGE   500000

#define NTILES   (NEDGE / 16)     // 31250 edge tiles of 16
#define NPAIRS   (NTILES / 2)     // 15625 tile pairs (32 edges each)
#define NT       13               // n-tiles: ceil(200/16) -> 208 padded cols
#define KSTEPS   12               // 384 / 32
#define WA_ROWS  208              // 200 padded to 13*16
#define WA_STRIDE 392             // 384 + 8 halfs padding (bank-conflict-free ds_load_b128)
#define WAVES    8                // 256 threads
#define MAIN_GRID 256

typedef __attribute__((ext_vector_type(16))) _Float16 v16h;
typedef __attribute__((ext_vector_type(8)))  _Float16 v8h;
typedef __attribute__((ext_vector_type(4)))  _Float16 v4h;
typedef __attribute__((ext_vector_type(8)))  float    v8f;

union ABfrag { v16h v; v8h h[2]; };

// ---------------- zero init ----------------
__global__ void kg_zero_kernel(float* __restrict__ p, int n) {
    int i = blockIdx.x * blockDim.x + threadIdx.x;
    int stride = gridDim.x * blockDim.x;
    for (; i < n; i += stride) p[i] = 0.0f;
}

// ---------------- renorm + f32->f16 (one wave per 128-elem row) ----------------
__global__ void kg_renorm_kernel(const float* __restrict__ in,
                                 _Float16* __restrict__ outh, int rows) {
    int gtid = blockIdx.x * blockDim.x + threadIdx.x;
    int row  = gtid >> 5;
    int lane = gtid & 31;
    if (row >= rows) return;
    const float4* rp = (const float4*)(in + (size_t)row * IN_DIM);
    float4 v = rp[lane];
    float ss = v.x*v.x + v.y*v.y + v.z*v.z + v.w*v.w;
    #pragma unroll
    for (int m = 16; m >= 1; m >>= 1) ss += __shfl_xor(ss, m, 32);
    float nrm = sqrtf(ss);
    float sc  = fminf(1.0f, 1.0f / fmaxf(nrm, 1e-12f));
    v4h h;
    h[0] = (_Float16)(v.x * sc);
    h[1] = (_Float16)(v.y * sc);
    h[2] = (_Float16)(v.z * sc);
    h[3] = (_Float16)(v.w * sc);
    ((v4h*)(outh + (size_t)row * IN_DIM))[lane] = h;
}

// ---------------- main fused GEMM + attention + scatter ----------------
extern __shared__ char kg_smem[];

__device__ __forceinline__ void kg_loadB(ABfrag& f, const _Float16* lb, int t) {
    const _Float16* p = lb + t * 16 * WA_STRIDE;
    f.h[0] = *(const v8h*)(p);
    f.h[1] = *(const v8h*)(p + 16);
}

__global__ __attribute__((amdgpu_flat_work_group_size(256, 256),
                          amdgpu_waves_per_eu(2)))
void kg_main_kernel(
    const int*      __restrict__ trip,   // (E,3)
    const _Float16* __restrict__ entF,   // (N_ENT,128) renormed f16
    const _Float16* __restrict__ relF,   // (N_REL,128) renormed f16
    const float*    __restrict__ Wa,     // (200,384)
    const float*    __restrict__ ba,     // (200)
    const float*    __restrict__ Wa2,    // (200)
    const float*    __restrict__ ba2,    // (1)
    float*          __restrict__ outEnt, // d_out[0 .. N_ENT*200)  : hs accumulator
    float*          __restrict__ outRel, // d_out[N_ENT*200 ..)    : rel-sum accumulator
    float*          __restrict__ ebs,    // (N_ENT) attention denominators
    float*          __restrict__ cnt)    // (N_REL) edge counts
{
    _Float16* waLds   = (_Float16*)kg_smem;                          // 208*392 halfs
    float*    relpart = (float*)(kg_smem + WA_ROWS*WA_STRIDE*2);     // 200*200
    float*    cntpart = relpart + OUT_DIM*OUT_DIM;                   // 200
    float*    wscr    = cntpart + OUT_DIM;                           // 8*32

    const int tid = threadIdx.x;

    // --- stage Wa into LDS as f16, rows 200..207 zero-padded ---
    for (int i = tid; i < WA_ROWS * 48; i += 256) {
        int row = i / 48;
        int c8  = (i - row * 48) * 8;
        v8h h;
        if (row < OUT_DIM) {
            const float* wp = Wa + row * 384 + c8;
            #pragma unroll
            for (int q = 0; q < 8; ++q) h[q] = (_Float16)wp[q];
        } else {
            #pragma unroll
            for (int q = 0; q < 8; ++q) h[q] = (_Float16)0.0f;
        }
        *(v8h*)(waLds + row * WA_STRIDE + c8) = h;
    }
    for (int i = tid; i < OUT_DIM*OUT_DIM + OUT_DIM; i += 256) relpart[i] = 0.0f;
    __syncthreads();

    const int lane = tid & 31;
    const int j    = lane & 15;        // edge-in-tile / N-column lane
    const int sel  = (lane >> 4) & 1;  // K-half selector (ISA 16-bit A/B layout)
    const int wave = tid >> 5;

    // per-lane Wa2 / ba fragments (column n = 16t + j)
    float wa2v[NT], bav[NT];
    #pragma unroll
    for (int t = 0; t < NT; ++t) {
        int n = 16*t + j;
        wa2v[t] = (n < OUT_DIM) ? Wa2[n] : 0.0f;
        bav[t]  = (n < OUT_DIM) ? ba[n]  : 0.0f;
    }
    const float ba2s = ba2[0];
    float* wsc = wscr + wave * 32;

    // post-GEMM attention + scatter for one 16-edge tile
    auto kg_post = [&](v8f (&acc)[NT], int si, int ri, int wofs) {
        float pv[8];
        #pragma unroll
        for (int v = 0; v < 8; ++v) pv[v] = 0.0f;
        #pragma unroll
        for (int t = 0; t < NT; ++t) {
            #pragma unroll
            for (int v = 0; v < 8; ++v) {
                float c = acc[t][v] + bav[t];
                acc[t][v] = c;
                pv[v] += c * wa2v[t];
            }
        }
        #pragma unroll
        for (int m = 8; m >= 1; m >>= 1) {
            #pragma unroll
            for (int v = 0; v < 8; ++v) pv[v] += __shfl_xor(pv[v], m, 16);
        }
        float ebv[8];
        #pragma unroll
        for (int v = 0; v < 8; ++v) {
            float z  = pv[v] + ba2s;
            float lr = (z > 0.0f) ? z : 0.01f * z;
            ebv[v]   = __expf(-lr);
        }
        #pragma unroll
        for (int t = 0; t < NT; ++t) {
            #pragma unroll
            for (int v = 0; v < 8; ++v) acc[t][v] *= ebv[v];
        }
        int srow[8], rrow[8];
        #pragma unroll
        for (int v = 0; v < 8; ++v) {
            srow[v] = __shfl(si, v + 8*sel, 32);
            rrow[v] = __shfl(ri, v + 8*sel, 32);
        }
        #pragma unroll
        for (int t = 0; t < NT; ++t) {
            int n = 16*t + j;
            if (n < OUT_DIM) {
                #pragma unroll
                for (int v = 0; v < 8; ++v) {
                    float val = acc[t][v];
                    atomicAdd(outEnt + (size_t)srow[v] * OUT_DIM + n, val);
                    atomicAdd(relpart + rrow[v] * OUT_DIM + n, val);
                }
            }
        }
        if (lane == 0 || lane == 16) {
            #pragma unroll
            for (int v = 0; v < 8; ++v) wsc[wofs + sel*8 + v] = ebv[v];
        }
        float ebj = wsc[wofs + j];
        if (lane < 16) {
            atomicAdd(ebs + si, ebj);
            atomicAdd(cntpart + ri, 1.0f);
        }
    };

    for (int pair = blockIdx.x * WAVES + wave; pair < NPAIRS;
         pair += gridDim.x * WAVES) {
        int e0 = (pair * 2) * 16 + j;
        int e1 = e0 + 16;
        int si0 = trip[3*e0 + 0], ri0 = trip[3*e0 + 1], oi0 = trip[3*e0 + 2];
        int si1 = trip[3*e1 + 0], ri1 = trip[3*e1 + 1], oi1 = trip[3*e1 + 2];
        const _Float16* bs0 = entF + (size_t)si0 * IN_DIM;
        const _Float16* br0 = relF + (size_t)ri0 * IN_DIM;
        const _Float16* bo0 = entF + (size_t)oi0 * IN_DIM;
        const _Float16* bs1 = entF + (size_t)si1 * IN_DIM;
        const _Float16* br1 = relF + (size_t)ri1 * IN_DIM;
        const _Float16* bo1 = entF + (size_t)oi1 * IN_DIM;

        v8f acc0[NT] = {};
        v8f acc1[NT] = {};

        #pragma unroll
        for (int k = 0; k < KSTEPS; ++k) {
            const _Float16* rp0 = (k < 4) ? (bs0 + 32*k)
                                : (k < 8) ? (br0 + 32*k - 128)
                                          : (bo0 + 32*k - 256);
            const _Float16* rp1 = (k < 4) ? (bs1 + 32*k)
                                : (k < 8) ? (br1 + 32*k - 128)
                                          : (bo1 + 32*k - 256);
            // A fragments: lane sel=0 -> K{0..7,16..23}; sel=1 -> K{8..15,24..31}
            ABfrag af0, af1;
            af0.h[0] = *(const v8h*)(rp0 + sel*8);
            af0.h[1] = *(const v8h*)(rp0 + sel*8 + 16);
            af1.h[0] = *(const v8h*)(rp1 + sel*8);
            af1.h[1] = *(const v8h*)(rp1 + sel*8 + 16);

            // preload ALL 13 B fragments for this k-step into distinct buffers:
            // no WAR hazards with the WMMA chain, addresses computed once,
            // loads clause together and overlap the 26-WMMA chain.
            const _Float16* lb = waLds + j * WA_STRIDE + 32*k + sel*8;
            ABfrag bf[NT];
            #pragma unroll
            for (int t = 0; t < NT; ++t) kg_loadB(bf[t], lb, t);

            #pragma unroll
            for (int t = 0; t < NT; ++t) {
                acc0[t] = __builtin_amdgcn_wmma_f32_16x16x32_f16(
                    false, af0.v, false, bf[t].v, (short)0, acc0[t], false, false);
                acc1[t] = __builtin_amdgcn_wmma_f32_16x16x32_f16(
                    false, af1.v, false, bf[t].v, (short)0, acc1[t], false, false);
            }
        }

        kg_post(acc0, si0, ri0, 0);
        kg_post(acc1, si1, ri1, 16);
    }

    __syncthreads();
    // flush relation partials + counts
    for (int i = tid; i < OUT_DIM*OUT_DIM; i += 256)
        atomicAdd(outRel + i, relpart[i]);
    for (int i = tid; i < OUT_DIM; i += 256)
        atomicAdd(cnt + i, cntpart[i]);
}

// ---------------- finalize ----------------
__global__ void kg_fin_ent_kernel(float* __restrict__ out,
                                  const float* __restrict__ ebs, int n) {
    int i = blockIdx.x * blockDim.x + threadIdx.x;
    if (i >= n) return;
    int row = i / OUT_DIM;
    float d = ebs[row];
    d = (d == 0.0f) ? 1e-12f : d;
    float v = out[i] / d;
    out[i] = fmaxf(v, 0.0f);
}

__global__ void kg_fin_rel_kernel(float* __restrict__ outRel,
                                  const float* __restrict__ cnt, int n) {
    int i = blockIdx.x * blockDim.x + threadIdx.x;
    if (i >= n) return;
    int row = i / OUT_DIM;
    float d = fmaxf(cnt[row], 1.0f);
    float v = outRel[i] / d;
    outRel[i] = fmaxf(v, 0.0f);
}

// ---------------- host launcher ----------------
extern "C" void kernel_launch(void* const* d_in, const int* in_sizes, int n_in,
                              void* d_out, int out_size, void* d_ws, size_t ws_size,
                              hipStream_t stream) {
    const int*   trip = (const int*)  d_in[0];
    const float* ent  = (const float*)d_in[1];
    const float* rel  = (const float*)d_in[2];
    const float* Wa   = (const float*)d_in[3];
    const float* ba   = (const float*)d_in[4];
    const float* Wa2  = (const float*)d_in[5];
    const float* ba2  = (const float*)d_in[6];
    float* out = (float*)d_out;

    // workspace layout
    _Float16* entF = (_Float16*)d_ws;                                   // N_ENT*128
    _Float16* relF = entF + (size_t)N_ENT * IN_DIM;                     // N_REL*128
    float*    ebs  = (float*)(relF + (size_t)N_REL * IN_DIM);           // N_ENT
    float*    cnt  = ebs + N_ENT;                                       // N_REL

    const int totalOut = N_ENT * OUT_DIM + N_REL * OUT_DIM;             // 20,040,000

    // zero accumulators (output doubles as hs / rel-sum accumulator)
    kg_zero_kernel<<<2048, 256, 0, stream>>>(out, totalOut);
    kg_zero_kernel<<<128, 256, 0, stream>>>(ebs, N_ENT + N_REL);

    // renorm embeddings -> f16
    kg_renorm_kernel<<<(N_ENT * 32 + 255) / 256, 256, 0, stream>>>(ent, entF, N_ENT);
    kg_renorm_kernel<<<(N_REL * 32 + 255) / 256, 256, 0, stream>>>(rel, relF, N_REL);

    // fused GEMM + attention + scatter
    size_t smem = (size_t)WA_ROWS * WA_STRIDE * 2       // Wa f16 tile
                + (size_t)OUT_DIM * OUT_DIM * 4         // relation partials
                + (size_t)OUT_DIM * 4                   // count partials
                + (size_t)WAVES * 32 * 4;               // wave scratch
    kg_main_kernel<<<MAIN_GRID, 256, smem, stream>>>(
        trip, entF, relF, Wa, ba, Wa2, ba2,
        out, out + (size_t)N_ENT * OUT_DIM, ebs, cnt);

    // finalize
    kg_fin_ent_kernel<<<(N_ENT * OUT_DIM + 255) / 256, 256, 0, stream>>>(out, ebs, N_ENT * OUT_DIM);
    kg_fin_rel_kernel<<<(N_REL * OUT_DIM + 255) / 256, 256, 0, stream>>>(
        out + (size_t)N_ENT * OUT_DIM, cnt, N_REL * OUT_DIM);
}
